// MultilayerGRU_39505109188955
// MI455X (gfx1250) — matrix operand order repaired and checked by reading, not compiled, not run on hardware
//
#include <hip/hip_runtime.h>

// ---------------------------------------------------------------------------
// Multilayer GRU on MI455X (gfx1250):
//   persistent kernel, device-wide sense-reversing barrier, bf16 WMMA
//   (v_wmma_f32_16x16x32_bf16) with fp32 accumulation and fp32 state.
// ---------------------------------------------------------------------------

#define IN_DIM  512
#define H_DIM   1024
#define OUT_DIM 512
#define BATCH   64
#define SEQ     512
#define NLAY    2

#define NBLK 64
#define TPB  256
#define NWAVES (NBLK * (TPB / 32))   // 512 wave32s

typedef __attribute__((ext_vector_type(16))) __bf16 v16bf;
typedef __attribute__((ext_vector_type(8)))  float  v8f;
typedef unsigned short ushort_t;

// fp32 -> bf16 round-to-nearest-even (bit trick; NaN edge ignored)
__device__ __forceinline__ ushort_t f2bf(float f) {
  unsigned int u = __float_as_uint(f);
  unsigned int r = u + 0x7fffu + ((u >> 16) & 1u);
  return (ushort_t)(r >> 16);
}

__device__ __forceinline__ float sigmoid_f(float x) {
  return 1.0f / (1.0f + __expf(-x));
}
__device__ __forceinline__ float tanh_f(float x) {
  float e = __expf(-2.0f * fabsf(x));
  float t = (1.0f - e) / (1.0f + e);
  return copysignf(t, x);
}

union BFrag { v16bf v; uint4 q[2]; };

// Accumulate a K-span of D += A(16 x K) * B^T (B is row-major N x K weights).
//   A : bf16 row-major, caller pre-offset to tile row 0; lda = row stride.
//   B : bf16 row-major (N x K) weights, caller pre-offset to tile col row 0
//       and K offset; ldb = row stride.
// A-frag layout (ISA 7.12.2, 16-bit A 16x32): lane l -> M = l&15,
//   lanes 0-15 hold K {k..k+7, k+16..k+23}, lanes 16-31 hold {k+8.., k+24..}.
// B-frag layout: lane l -> N = l&15, same K split as A: lanes 0-15 hold K {k..k+7, k+16..k+23}, lanes 16-31 hold {k+8.., k+24..}.
__device__ __forceinline__ void mma_span(
    v8f& acc,
    const ushort_t* __restrict__ A, long lda,
    const ushort_t* __restrict__ B, long ldb,
    int kCount, int lane) {
  const int hh  = lane >> 4;
  const int l15 = lane & 15;
  const ushort_t* arow = A + (long)l15 * lda + hh * 8;
  const ushort_t* brow = B + (long)l15 * ldb + hh * 8;
  for (int k = 0; k < kCount; k += 32) {
    BFrag a, b;
    a.q[0] = *(const uint4*)(arow + k);
    a.q[1] = *(const uint4*)(arow + k + 16);
    b.q[0] = *(const uint4*)(brow + k);
    b.q[1] = *(const uint4*)(brow + k + 16);
    acc = __builtin_amdgcn_wmma_f32_16x16x32_bf16(
        false, a.v, false, b.v, (short)0, acc, false, false);
  }
}

struct P {
  const ushort_t* xb;                       // (B,S,IN) bf16
  const ushort_t* Wz0; const ushort_t* Wr0; const ushort_t* Wg0;  // (H, IN+H)
  const ushort_t* Wz1; const ushort_t* Wr1; const ushort_t* Wg1;  // (H, 2H)
  const ushort_t* Wyb;                      // (OUT, H)
  const float* bz0; const float* br0; const float* bg0;
  const float* bz1; const float* br1; const float* bg1;
  const float* by;
  float* h0; float* h1;                     // fp32 state (B,H)
  ushort_t* hb0; ushort_t* hb1;             // bf16 mirrors of h
  float* zbuf;                              // (B,H) z-gate scratch
  ushort_t* rhb;                            // (B,H) bf16 r*h scratch
  float* yout;                              // (B,S,OUT)
  float* hout;                              // (B,L,H)
  int* bcnt; int* bgen;                     // grid barrier state
};

__device__ __forceinline__ void grid_barrier(int* cnt, int* gen) {
  __threadfence();
  __syncthreads();
  if (threadIdx.x == 0) {
    int g = __hip_atomic_load(gen, __ATOMIC_RELAXED, __HIP_MEMORY_SCOPE_AGENT);
    int a = __hip_atomic_fetch_add(cnt, 1, __ATOMIC_ACQ_REL, __HIP_MEMORY_SCOPE_AGENT);
    if (a == NBLK - 1) {
      __hip_atomic_store(cnt, 0, __ATOMIC_RELAXED, __HIP_MEMORY_SCOPE_AGENT);
      __hip_atomic_fetch_add(gen, 1, __ATOMIC_RELEASE, __HIP_MEMORY_SCOPE_AGENT);
    } else {
      while (__hip_atomic_load(gen, __ATOMIC_ACQUIRE, __HIP_MEMORY_SCOPE_AGENT) == g) {
        __builtin_amdgcn_s_sleep(1);
      }
    }
  }
  __syncthreads();
  __threadfence();
}

// One 16x16 tile of y_t = hb1 @ Wy^T + by
__device__ __forceinline__ void ytile(const P& p, int tile, int t, int lane) {
  const int l15 = lane & 15, hh = lane >> 4;
  const int mB = (tile >> 5) << 4;   // 4 m-tiles (B=64)
  const int nB = (tile & 31) << 4;   // 32 n-tiles (OUT=512)
  float bv = p.by[nB + l15];
  v8f acc = {bv, bv, bv, bv, bv, bv, bv, bv};
  mma_span(acc, p.hb1 + (long)mB * H_DIM, H_DIM,
           p.Wyb + (long)nB * H_DIM, H_DIM, H_DIM, lane);
#pragma unroll
  for (int v = 0; v < 8; ++v) {
    int m = mB + v + hh * 8;
    p.yout[((long)m * SEQ + t) * OUT_DIM + nB + l15] = acc[v];
  }
}

__global__ __launch_bounds__(TPB) void gru_persistent(P p) {
  const int lane = threadIdx.x & 31;
  const int wave = blockIdx.x * (TPB / 32) + (threadIdx.x >> 5);
  const int l15 = lane & 15, hh = lane >> 4;

  for (int t = 0; t < SEQ; ++t) {
    for (int layer = 0; layer < NLAY; ++layer) {
      const ushort_t* inp = layer ? p.hb0 : (p.xb + (long)t * IN_DIM);
      const long inp_ld   = layer ? (long)H_DIM : (long)SEQ * IN_DIM;
      const int  Kin      = layer ? H_DIM : IN_DIM;
      const long KW       = Kin + H_DIM;
      const ushort_t* Wz = layer ? p.Wz1 : p.Wz0;
      const ushort_t* Wr = layer ? p.Wr1 : p.Wr0;
      const ushort_t* Wg = layer ? p.Wg1 : p.Wg0;
      const float* bz = layer ? p.bz1 : p.bz0;
      const float* br = layer ? p.br1 : p.br0;
      const float* bg = layer ? p.bg1 : p.bg0;
      float*    h  = layer ? p.h1  : p.h0;
      ushort_t* hb = layer ? p.hb1 : p.hb0;

      // ---- phase A: z and r tiles (512 tiles over 512 waves) ----
      {
        const int gate = wave >> 8;            // 0 = z, 1 = r
        const int tile = wave & 255;
        const int mB = (tile >> 6) << 4;
        const int nB = (tile & 63) << 4;
        const ushort_t* W = gate ? Wr : Wz;
        const float* bias = gate ? br : bz;
        float bv = bias[nB + l15];
        v8f acc = {bv, bv, bv, bv, bv, bv, bv, bv};
        mma_span(acc, inp + (long)mB * inp_ld, inp_ld,
                 W + (long)nB * KW, KW, Kin, lane);
        mma_span(acc, hb + (long)mB * H_DIM, H_DIM,
                 W + (long)nB * KW + Kin, KW, H_DIM, lane);
#pragma unroll
        for (int v = 0; v < 8; ++v) {
          int idx = (mB + v + hh * 8) * H_DIM + nB + l15;
          float s = sigmoid_f(acc[v]);
          if (gate == 0) p.zbuf[idx] = s;
          else           p.rhb[idx]  = f2bf(s * h[idx]);
        }
      }
      grid_barrier(p.bcnt, p.bgen);

      // ---- phase B: g tiles + state update (waves 0..255);
      //      waves 256..383 do y(t-1) during layer 0 ----
      if (wave < 256) {
        const int mB = (wave >> 6) << 4;
        const int nB = (wave & 63) << 4;
        float bv = bg[nB + l15];
        v8f acc = {bv, bv, bv, bv, bv, bv, bv, bv};
        mma_span(acc, inp + (long)mB * inp_ld, inp_ld,
                 Wg + (long)nB * KW, KW, Kin, lane);
        mma_span(acc, p.rhb + (long)mB * H_DIM, H_DIM,
                 Wg + (long)nB * KW + Kin, KW, H_DIM, lane);
#pragma unroll
        for (int v = 0; v < 8; ++v) {
          int idx = (mB + v + hh * 8) * H_DIM + nB + l15;
          float g  = tanh_f(acc[v]);
          float z  = p.zbuf[idx];
          float hn = h[idx] * z + (1.0f - z) * g;
          h[idx]  = hn;
          hb[idx] = f2bf(hn);
        }
      } else if (layer == 0 && wave < 384 && t > 0) {
        ytile(p, wave - 256, t - 1, lane);   // hb1 stable until layer-1 phase B
      }
      grid_barrier(p.bcnt, p.bgen);
    }
  }

  // ---- tail: y(SEQ-1) + final hidden state ----
  if (wave < 128) {
    ytile(p, wave, SEQ - 1, lane);
  } else {
    int tid = (wave - 128) * 32 + lane;
    const int total = NLAY * BATCH * H_DIM;
    for (int i = tid; i < total; i += (NWAVES - 128) * 32) {
      int l   = i / (BATCH * H_DIM);
      int rem = i - l * BATCH * H_DIM;
      int b   = rem / H_DIM;
      int hx  = rem - b * H_DIM;
      const float* hs = l ? p.h1 : p.h0;
      p.hout[(b * NLAY + l) * H_DIM + hx] = hs[rem];
    }
  }
}

// ------------------------- prologue kernels -------------------------------

__global__ void k_cvt_bf16(const float* __restrict__ src,
                           ushort_t* __restrict__ dst, long n) {
  long i  = (long)blockIdx.x * blockDim.x + threadIdx.x;
  long st = (long)gridDim.x * blockDim.x;
  for (; i < n; i += st) dst[i] = f2bf(src[i]);
}

// dst (H, Kin+H) bf16 = [Wx | Wh] row-concat
__global__ void k_wcat(const float* __restrict__ Wx, const float* __restrict__ Wh,
                       ushort_t* __restrict__ dst, int Kin) {
  const long KW = Kin + H_DIM;
  const long n  = (long)H_DIM * KW;
  long i  = (long)blockIdx.x * blockDim.x + threadIdx.x;
  long st = (long)gridDim.x * blockDim.x;
  for (; i < n; i += st) {
    int row = (int)(i / KW);
    int k   = (int)(i - (long)row * KW);
    float v = (k < Kin) ? Wx[(long)row * Kin + k]
                        : Wh[(long)row * H_DIM + (k - Kin)];
    dst[i] = f2bf(v);
  }
}

__global__ void k_init(const float* __restrict__ hidden, float* h0, float* h1,
                       ushort_t* hb0, ushort_t* hb1, int* bcnt, int* bgen) {
  int i  = blockIdx.x * blockDim.x + threadIdx.x;
  int st = gridDim.x * blockDim.x;
  if (i == 0) { *bcnt = 0; *bgen = 0; }
  for (int j = i; j < BATCH * NLAY * H_DIM; j += st) {
    int b   = j / (NLAY * H_DIM);
    int rem = j - b * NLAY * H_DIM;
    int l   = rem / H_DIM;
    int hx  = rem - l * H_DIM;
    float v = hidden[j];
    if (l == 0) { h0[b * H_DIM + hx] = v; hb0[b * H_DIM + hx] = f2bf(v); }
    else        { h1[b * H_DIM + hx] = v; hb1[b * H_DIM + hx] = f2bf(v); }
  }
}

// ------------------------------ launch ------------------------------------

extern "C" void kernel_launch(void* const* d_in, const int* in_sizes, int n_in,
                              void* d_out, int out_size, void* d_ws, size_t ws_size,
                              hipStream_t stream) {
  (void)in_sizes; (void)n_in; (void)out_size; (void)ws_size;

  const float* x    = (const float*)d_in[0];
  const float* hid  = (const float*)d_in[1];
  // layer 0 params (dict order: Wxz,Wxr,Wxg,Whz,bhz,Whr,bhr,Whg,bhg)
  const float* Wxz0 = (const float*)d_in[2];
  const float* Wxr0 = (const float*)d_in[3];
  const float* Wxg0 = (const float*)d_in[4];
  const float* Whz0 = (const float*)d_in[5];
  const float* bhz0 = (const float*)d_in[6];
  const float* Whr0 = (const float*)d_in[7];
  const float* bhr0 = (const float*)d_in[8];
  const float* Whg0 = (const float*)d_in[9];
  const float* bhg0 = (const float*)d_in[10];
  // layer 1
  const float* Wxz1 = (const float*)d_in[11];
  const float* Wxr1 = (const float*)d_in[12];
  const float* Wxg1 = (const float*)d_in[13];
  const float* Whz1 = (const float*)d_in[14];
  const float* bhz1 = (const float*)d_in[15];
  const float* Whr1 = (const float*)d_in[16];
  const float* bhr1 = (const float*)d_in[17];
  const float* Whg1 = (const float*)d_in[18];
  const float* bhg1 = (const float*)d_in[19];
  const float* Wy   = (const float*)d_in[20];
  const float* by   = (const float*)d_in[21];

  char* ws = (char*)d_ws;
  size_t off = 0;
  auto take = [&](size_t bytes) -> char* {
    char* q = ws + off;
    off += (bytes + 255) & ~(size_t)255;
    return q;
  };

  int*      bcnt = (int*)take(256);           // bcnt at +0, bgen at +128
  int*      bgen = (int*)((char*)bcnt + 128);
  ushort_t* xb   = (ushort_t*)take((size_t)BATCH * SEQ * IN_DIM * 2);
  ushort_t* Wz0  = (ushort_t*)take((size_t)H_DIM * (IN_DIM + H_DIM) * 2);
  ushort_t* Wr0  = (ushort_t*)take((size_t)H_DIM * (IN_DIM + H_DIM) * 2);
  ushort_t* Wg0  = (ushort_t*)take((size_t)H_DIM * (IN_DIM + H_DIM) * 2);
  ushort_t* Wz1  = (ushort_t*)take((size_t)H_DIM * (2 * H_DIM) * 2);
  ushort_t* Wr1  = (ushort_t*)take((size_t)H_DIM * (2 * H_DIM) * 2);
  ushort_t* Wg1  = (ushort_t*)take((size_t)H_DIM * (2 * H_DIM) * 2);
  ushort_t* Wyb  = (ushort_t*)take((size_t)OUT_DIM * H_DIM * 2);
  float*    h0   = (float*)take((size_t)BATCH * H_DIM * 4);
  float*    h1   = (float*)take((size_t)BATCH * H_DIM * 4);
  ushort_t* hb0  = (ushort_t*)take((size_t)BATCH * H_DIM * 2);
  ushort_t* hb1  = (ushort_t*)take((size_t)BATCH * H_DIM * 2);
  float*    zbuf = (float*)take((size_t)BATCH * H_DIM * 4);
  ushort_t* rhb  = (ushort_t*)take((size_t)BATCH * H_DIM * 2);

  // Prologue: bf16 conversions + [Wx|Wh] concatenation + state/barrier init.
  k_cvt_bf16<<<2048, 256, 0, stream>>>(x, xb, (long)BATCH * SEQ * IN_DIM);
  k_wcat<<<768, 256, 0, stream>>>(Wxz0, Whz0, Wz0, IN_DIM);
  k_wcat<<<768, 256, 0, stream>>>(Wxr0, Whr0, Wr0, IN_DIM);
  k_wcat<<<768, 256, 0, stream>>>(Wxg0, Whg0, Wg0, IN_DIM);
  k_wcat<<<1024, 256, 0, stream>>>(Wxz1, Whz1, Wz1, H_DIM);
  k_wcat<<<1024, 256, 0, stream>>>(Wxr1, Whr1, Wr1, H_DIM);
  k_wcat<<<1024, 256, 0, stream>>>(Wxg1, Whg1, Wg1, H_DIM);
  k_cvt_bf16<<<512, 256, 0, stream>>>(Wy, Wyb, (long)OUT_DIM * H_DIM);
  k_init<<<64, 256, 0, stream>>>(hid, h0, h1, hb0, hb1, bcnt, bgen);

  P prm;
  prm.xb = xb;
  prm.Wz0 = Wz0; prm.Wr0 = Wr0; prm.Wg0 = Wg0;
  prm.Wz1 = Wz1; prm.Wr1 = Wr1; prm.Wg1 = Wg1;
  prm.Wyb = Wyb;
  prm.bz0 = bhz0; prm.br0 = bhr0; prm.bg0 = bhg0;
  prm.bz1 = bhz1; prm.br1 = bhr1; prm.bg1 = bhg1;
  prm.by = by;
  prm.h0 = h0; prm.h1 = h1; prm.hb0 = hb0; prm.hb1 = hb1;
  prm.zbuf = zbuf; prm.rhb = rhb;
  prm.yout = (float*)d_out;
  prm.hout = (float*)d_out + (size_t)BATCH * SEQ * OUT_DIM;
  prm.bcnt = bcnt; prm.bgen = bgen;

  gru_persistent<<<NBLK, TPB, 0, stream>>>(prm);
}
